// FullResAttnDFCBlock_36335423324244
// MI455X (gfx1250) — matrix operand order repaired and checked by reading, not compiled, not run on hardware
//
#include <hip/hip_runtime.h>
#include <hip/hip_bf16.h>

typedef __attribute__((ext_vector_type(16))) _Float16 v16h;
typedef __attribute__((ext_vector_type(8)))  _Float16 v8h;
typedef __attribute__((ext_vector_type(8)))  float    v8f;

// async global->LDS b64 moves 8 bytes; toolchain declares the operands as
// pointers to int __attribute__((vector_size(8))) in AS1 (global) / AS3 (LDS).
typedef int v2i __attribute__((vector_size(2 * sizeof(int))));
typedef __attribute__((address_space(1))) v2i* gv2i_p;
typedef __attribute__((address_space(3))) v2i* lv2i_p;

#define EPS_BN 1e-5f

constexpr int B_  = 4;
constexpr int CI  = 64;
constexpr int CO  = 128;
constexpr int H_  = 64;
constexpr int W_  = 64;
constexpr int NP  = H_ * W_;   // 4096 pixels
constexpr int CQ  = 16;        // q/k channels
constexpr int PPB = 8;         // pixels per block in projection/fusion kernels

__device__ __forceinline__ float bn_f(float x, float g, float b, float m, float v) {
  return (x - m) * rsqrtf(v + EPS_BN) * g + b;
}

__device__ __forceinline__ v8f v8f_zero() {
  v8f z;
#pragma unroll
  for (int i = 0; i < 8; ++i) z[i] = 0.0f;
  return z;
}

__device__ __forceinline__ v16h v16h_zero() {
  v16h z;
#pragma unroll
  for (int i = 0; i < 16; ++i) z[i] = (_Float16)0.0f;
  return z;
}

// DPP row-rotate (16-lane row) for cross-lane reductions on the VALU —
// avoids ds_bpermute in the flash-attention inner loop. ROW_ROR:n = 0x120|n.
template <int CTRL>
__device__ __forceinline__ float dpp_rotf(float x) {
  return __int_as_float(__builtin_amdgcn_update_dpp(
      0, __float_as_int(x), CTRL, 0xf, 0xf, true));
}

__device__ __forceinline__ float row16_max(float x) {
  x = fmaxf(x, dpp_rotf<0x121>(x));   // ror:1
  x = fmaxf(x, dpp_rotf<0x122>(x));   // ror:2
  x = fmaxf(x, dpp_rotf<0x124>(x));   // ror:4
  x = fmaxf(x, dpp_rotf<0x128>(x));   // ror:8
  return x;
}

__device__ __forceinline__ float row16_sum(float x) {
  x += dpp_rotf<0x121>(x);
  x += dpp_rotf<0x122>(x);
  x += dpp_rotf<0x124>(x);
  x += dpp_rotf<0x128>(x);
  return x;
}

// ---------------------------------------------------------------------------
// 3x3 conv (pad=1) + BN + ReLU -> local [B,CO,NP]
// one block per (b, h, 32-channel chunk); 64 threads = w.
// x rows are staged once per ci and reused by 32 output channels (32x less
// x traffic than one-channel-per-block).
// ---------------------------------------------------------------------------
__global__ __launch_bounds__(64) void conv3_bn_relu(
    const float* __restrict__ x, const float* __restrict__ W3,
    const float* __restrict__ b3, const float* __restrict__ g1,
    const float* __restrict__ b1, const float* __restrict__ m1,
    const float* __restrict__ v1, float* __restrict__ local_out) {
  __shared__ float xs[3][W_ + 2];
  const int w   = threadIdx.x;
  const int coc = blockIdx.x % (CO / 32);
  const int h   = (blockIdx.x / (CO / 32)) % H_;
  const int b   = blockIdx.x / ((CO / 32) * H_);
  const int co0 = coc * 32;

  if (w == 0)      { xs[0][0] = 0.f; xs[1][0] = 0.f; xs[2][0] = 0.f; }
  if (w == W_ - 1) { xs[0][W_ + 1] = 0.f; xs[1][W_ + 1] = 0.f; xs[2][W_ + 1] = 0.f; }

  float acc[32];
#pragma unroll
  for (int j = 0; j < 32; ++j) acc[j] = 0.f;

  const float* xb = x + (size_t)b * CI * NP;

  for (int ci = 0; ci < CI; ++ci) {
#pragma unroll
    for (int hr = 0; hr < 3; ++hr) {
      int hh = h + hr - 1;
      xs[hr][w + 1] = (hh >= 0 && hh < H_) ? xb[ci * NP + hh * W_ + w] : 0.f;
    }
    __syncthreads();
    for (int j = 0; j < 32; ++j) {
      const float* wk = W3 + ((size_t)(co0 + j) * CI + ci) * 9;  // uniform
#pragma unroll
      for (int kh = 0; kh < 3; ++kh)
#pragma unroll
        for (int kw = 0; kw < 3; ++kw)
          acc[j] = fmaf(wk[kh * 3 + kw], xs[kh][w + kw], acc[j]);
    }
    __syncthreads();
  }
#pragma unroll 4
  for (int j = 0; j < 32; ++j) {
    const int co = co0 + j;
    float y = bn_f(acc[j] + b3[co], g1[co], b1[co], m1[co], v1[co]);
    local_out[((size_t)b * CO + co) * NP + h * W_ + w] = fmaxf(y, 0.f);
  }
}

// ---------------------------------------------------------------------------
// a = ReLU(BN(Wa x + ba)); 8 pixels per block so each Wa element is read once
// per 8 outputs. 128 threads = out channel.
// ---------------------------------------------------------------------------
__global__ __launch_bounds__(CO) void a_proj(
    const float* __restrict__ x, const float* __restrict__ Wa,
    const float* __restrict__ ba, const float* __restrict__ g2,
    const float* __restrict__ b2, const float* __restrict__ m2,
    const float* __restrict__ v2, float* __restrict__ a_out) {
  __shared__ float xv[CI][PPB];
  const int c    = threadIdx.x;
  const int pix0 = (blockIdx.x % (NP / PPB)) * PPB;
  const int b    = blockIdx.x / (NP / PPB);

  for (int idx = c; idx < CI * PPB; idx += CO)
    xv[idx >> 3][idx & 7] = x[((size_t)b * CI + (idx >> 3)) * NP + pix0 + (idx & 7)];
  __syncthreads();

  float acc[PPB];
#pragma unroll
  for (int p = 0; p < PPB; ++p) acc[p] = ba[c];
  const float* wr = Wa + (size_t)c * CI;
  for (int i = 0; i < CI; ++i) {
    const float wvv = wr[i];
#pragma unroll
    for (int p = 0; p < PPB; ++p) acc[p] = fmaf(wvv, xv[i][p], acc[p]);
  }
  const float gg = g2[c], bb = b2[c], mm = m2[c], vv = v2[c];
#pragma unroll
  for (int p = 0; p < PPB; ++p)
    a_out[((size_t)b * CO + c) * NP + pix0 + p] = fmaxf(bn_f(acc[p], gg, bb, mm, vv), 0.f);
}

// ---------------------------------------------------------------------------
// q/k/v projections from a (f16 outputs); 8 pixels per block, 160 threads
// (5 waves): 16 q + 16 k + 128 v channels.
// ---------------------------------------------------------------------------
__global__ __launch_bounds__(160) void qkv_proj(
    const float* __restrict__ a,
    const float* __restrict__ Wq, const float* __restrict__ bq,
    const float* __restrict__ Wk, const float* __restrict__ bk,
    const float* __restrict__ Wv, const float* __restrict__ bv,
    _Float16* __restrict__ qh, _Float16* __restrict__ kh,
    _Float16* __restrict__ vh) {
  __shared__ float av[CO][PPB];
  const int t    = threadIdx.x;
  const int pix0 = (blockIdx.x % (NP / PPB)) * PPB;
  const int b    = blockIdx.x / (NP / PPB);

  for (int idx = t; idx < CO * PPB; idx += 160)
    av[idx >> 3][idx & 7] = a[((size_t)b * CO + (idx >> 3)) * NP + pix0 + (idx & 7)];
  __syncthreads();

  const float* wr;
  _Float16* dst;
  float bias;
  if (t < CQ) {
    wr = Wq + (size_t)t * CO;          bias = bq[t];
    dst = qh + ((size_t)b * CQ + t) * NP + pix0;
  } else if (t < 2 * CQ) {
    wr = Wk + (size_t)(t - CQ) * CO;   bias = bk[t - CQ];
    dst = kh + ((size_t)b * CQ + (t - CQ)) * NP + pix0;
  } else {
    wr = Wv + (size_t)(t - 2 * CQ) * CO; bias = bv[t - 2 * CQ];
    dst = vh + ((size_t)b * CO + (t - 2 * CQ)) * NP + pix0;
  }
  float acc[PPB];
#pragma unroll
  for (int p = 0; p < PPB; ++p) acc[p] = bias;
  for (int i = 0; i < CO; ++i) {
    const float wvv = wr[i];
#pragma unroll
    for (int p = 0; p < PPB; ++p) acc[p] = fmaf(wvv, av[i][p], acc[p]);
  }
#pragma unroll
  for (int p = 0; p < PPB; ++p) dst[p] = (_Float16)acc[p];
}

// ---------------------------------------------------------------------------
// Flash attention, never materializes the 4096x4096 matrix.
// Block = 256 threads (8 waves); each wave owns 16 query rows; block owns a
// 128-query strip of one batch. Stream over 32-key blocks with DOUBLE-BUFFERED
// async global->LDS staging (issue block i+1, s_wait_asynccnt 4 so block i's
// 4 per-wave transfers are complete while the next 4 stay in flight):
//   S[16x32]  = Qt[16x16(pad32)] x K[16(pad32)x32]   (2 wmma f32_16x16x32_f16)
//   online softmax (DPP ROW_ROR rotate-reduce, rescale O)
//   O[16x128] += P[16x32] x Vt[32x128]               (8 wmma, K=32 exact)
// ---------------------------------------------------------------------------
__global__ __launch_bounds__(256) void attn_flash(
    const _Float16* __restrict__ qg, const _Float16* __restrict__ kg,
    const _Float16* __restrict__ vg, float* __restrict__ og) {
  __shared__ alignas(64) _Float16 Kt[2][32][16];   // [buf][key m][cq]
  __shared__ alignas(64) _Float16 Vt[2][CO][32];   // [buf][channel][key m]
  __shared__ alignas(64) _Float16 Pl[8][16][32];   // per-wave P scratch
  __shared__ alignas(64) _Float16 Kz[16];          // zero row (hi-lane B-frag)

  const int tid  = threadIdx.x;
  const int lane = tid & 31;
  const int wv   = tid >> 5;
  const int l16  = lane & 15;
  const bool hi  = lane >= 16;

  const int b  = blockIdx.x >> 5;          // 32 query blocks per batch
  const int rb = blockIdx.x & 31;
  const int n0 = rb * 128 + wv * 16;       // this wave's query strip

  const _Float16* qb = qg + (size_t)b * CQ * NP;
  const _Float16* kb = kg + (size_t)b * CQ * NP;
  const _Float16* vb = vg + (size_t)b * CO * NP;

  if (tid < 16) Kz[tid] = (_Float16)0.0f;  // persistent zero region

  // cooperative staging of one 32-key block into buffer `buf`
  auto stage = [&](int buf, int m0) {
    for (int idx = tid; idx < 32 * 16; idx += 256) {       // K^T, small gather
      const int mm = idx >> 4, cc = idx & 15;
      Kt[buf][mm][cc] = kb[cc * NP + m0 + mm];
    }
    for (int idx = tid; idx < CO * 8; idx += 256) {        // V: 4 async b64/wave
      const int cc = idx >> 3, mm4 = (idx & 7) * 4;
      __builtin_amdgcn_global_load_async_to_lds_b64(
          (gv2i_p)(vb + (size_t)cc * NP + m0 + mm4),
          (lv2i_p)&Vt[buf][cc][mm4], 0, 0);
    }
  };

  // A-fragment of Q^T (16 queries x K=32; channels 16..31 are zero padding).
  // A frag, contiguous k order used here: lanes 0-15 row M=lane  h0-15->K=0..15
  //                  lanes16-31 row M=l-16 h0-15->K=16..31
  v16h qa = v16h_zero();
  {
    const int n = n0 + l16;
    if (!hi) {
#pragma unroll
      for (int h = 0; h < 16; ++h) qa[h] = qb[h * NP + n];
    }
    // hi lanes stay zero (K >= 16 padding)
  }

  v8f o[8];
#pragma unroll
  for (int t = 0; t < 8; ++t) o[t] = v8f_zero();
  float run_max[8], run_sum[8];
#pragma unroll
  for (int r = 0; r < 8; ++r) { run_max[r] = -INFINITY; run_sum[r] = 0.f; }

  stage(0, 0);                               // prologue: fill buffer 0

  constexpr int NBLK = NP / 32;              // 128 key blocks
  for (int i = 0; i < NBLK; ++i) {
    const int cur = i & 1;
    if (i + 1 < NBLK) {
      stage(cur ^ 1, (i + 1) * 32);          // prefetch next block (no wait)
      asm volatile("s_wait_asynccnt 4" ::: "memory");  // cur's 4 done, next in flight
    } else {
      asm volatile("s_wait_asynccnt 0" ::: "memory");
    }
    __syncthreads();                         // all waves' cur staging visible

    // ---- S = Q^T K : two 16x16 output tiles over the 32 keys -------------
    // 16-bit B layout (32x16): col = lane%16, lanes 0-15 rows K=0..15 (ctg),
    // lanes 16-31 rows K=16..31 -> shared zero row (channel padding).
    const _Float16* k0p = hi ? &Kz[0] : &Kt[cur][l16][0];
    const _Float16* k1p = hi ? &Kz[0] : &Kt[cur][16 + l16][0];
    const v16h kf0 = *(const v16h*)k0p;
    const v16h kf1 = *(const v16h*)k1p;
    const v8f cz = v8f_zero();
    v8f s0 = __builtin_amdgcn_wmma_f32_16x16x32_f16(false, qa, false, kf0,
                                                    (short)0, cz, false, false);
    v8f s1 = __builtin_amdgcn_wmma_f32_16x16x32_f16(false, qa, false, kf1,
                                                    (short)0, cz, false, false);

    // ---- online softmax over the 32 new scores per query row ------------
    // C layout: VGPR r holds row M=r (lanes 0-15) / M=r+8 (lanes 16-31),
    // col = lane%16 -> row reductions are 16-lane DPP rotate-reductions.
#pragma unroll
    for (int r = 0; r < 8; ++r) {
      const float mx = row16_max(fmaxf(s0[r], s1[r]));
      const float nm   = fmaxf(run_max[r], mx);
      const float corr = __expf(run_max[r] - nm);
      const float p0 = __expf(s0[r] - nm);
      const float p1 = __expf(s1[r] - nm);
      const float sm = row16_sum(p0 + p1);
      run_sum[r] = run_sum[r] * corr + sm;
      run_max[r] = nm;
#pragma unroll
      for (int t = 0; t < 8; ++t) o[t][r] *= corr;   // rescale accumulators
      const int M = r + (hi ? 8 : 0);
      Pl[wv][M][l16]      = (_Float16)p0;            // C-layout -> LDS
      Pl[wv][M][16 + l16] = (_Float16)p1;
    }

    // re-read P in A-fragment layout (per-wave scratch, in-order DS ops)
    v16h pa;
    {
      const int M = l16;
      const int base8 = hi ? 16 : 0;
      const v8h plo = *(const v8h*)&Pl[wv][M][base8];
      const v8h phi = *(const v8h*)&Pl[wv][M][8 + base8];
#pragma unroll
      for (int h = 0; h < 8; ++h) { pa[h] = plo[h]; pa[8 + h] = phi[h]; }
    }

    // ---- O += P x V^T : 8 channel tiles, K=32 = one full key block -------
    // B layout: col = channel c0+lane%16; lanes 0-15 rows m=0..15 (ctg),
    // lanes 16-31 rows m=16..31 (ctg at offset 16).
#pragma unroll
    for (int t = 0; t < 8; ++t) {
      const v16h vf = *(const v16h*)&Vt[cur][t * 16 + l16][hi ? 16 : 0];
      o[t] = __builtin_amdgcn_wmma_f32_16x16x32_f16(false, pa, false, vf,
                                                    (short)0, o[t], false, false);
    }
    __syncthreads();                         // done reading cur before restage
  }

  // normalize and store o[b][c][n]
#pragma unroll
  for (int r = 0; r < 8; ++r) {
    const float inv = 1.f / run_sum[r];
    const int n = n0 + r + (hi ? 8 : 0);
#pragma unroll
    for (int t = 0; t < 8; ++t)
      og[((size_t)b * CO + (t * 16 + l16)) * NP + n] = o[t][r] * inv;
  }
}

// ---------------------------------------------------------------------------
// gated fusion + final projection + residual; 8 pixels per block so each
// Wg/Wf/Wr element is read once per 8 outputs (8x less weight traffic).
// ---------------------------------------------------------------------------
__global__ __launch_bounds__(CO) void fuse_out(
    const float* __restrict__ x, const float* __restrict__ local_in,
    const float* __restrict__ a_in, const float* __restrict__ o_in,
    const float* __restrict__ gam,
    const float* __restrict__ Wg, const float* __restrict__ bg,
    const float* __restrict__ g3, const float* __restrict__ b3n,
    const float* __restrict__ m3, const float* __restrict__ v3,
    const float* __restrict__ Wf, const float* __restrict__ bf,
    const float* __restrict__ g4, const float* __restrict__ b4,
    const float* __restrict__ m4, const float* __restrict__ v4,
    const float* __restrict__ Wr, const float* __restrict__ rs,
    float* __restrict__ out) {
  __shared__ float comb[2 * CO][PPB];   // [local ; attn_feat]
  __shared__ float fus[CO][PPB];
  __shared__ float xv[CI][PPB];
  const int c    = threadIdx.x;
  const int pix0 = (blockIdx.x % (NP / PPB)) * PPB;
  const int b    = blockIdx.x / (NP / PPB);
  const size_t base = ((size_t)b * CO + c) * NP + pix0;
  const float gamv = gam[0];

  float lc[PPB], at[PPB];
#pragma unroll
  for (int p = 0; p < PPB; ++p) {
    lc[p] = local_in[base + p];
    at[p] = gamv * o_in[base + p] + a_in[base + p];
    comb[c][p]      = lc[p];
    comb[CO + c][p] = at[p];
  }
  if (c < CI)
#pragma unroll
    for (int p = 0; p < PPB; ++p)
      xv[c][p] = x[((size_t)b * CI + c) * NP + pix0 + p];
  __syncthreads();

  float ga[PPB];
#pragma unroll
  for (int p = 0; p < PPB; ++p) ga[p] = bg[c];
  const float* wg = Wg + (size_t)c * 2 * CO;
  for (int j = 0; j < 2 * CO; ++j) {
    const float wvv = wg[j];
#pragma unroll
    for (int p = 0; p < PPB; ++p) ga[p] = fmaf(wvv, comb[j][p], ga[p]);
  }
  {
    const float gg = g3[c], bb = b3n[c], mm = m3[c], vv = v3[c];
#pragma unroll
    for (int p = 0; p < PPB; ++p) {
      const float gate = 1.f / (1.f + __expf(-bn_f(ga[p], gg, bb, mm, vv)));
      fus[c][p] = gate * lc[p] + (1.f - gate) * at[p];
    }
  }
  __syncthreads();

  float fa[PPB];
#pragma unroll
  for (int p = 0; p < PPB; ++p) fa[p] = bf[c];
  const float* wf = Wf + (size_t)c * 3 * CO;
  for (int j = 0; j < CO; ++j) {
    const float wvv = wf[j];
#pragma unroll
    for (int p = 0; p < PPB; ++p) fa[p] = fmaf(wvv, fus[j][p], fa[p]);
  }
  for (int j = 0; j < 2 * CO; ++j) {
    const float wvv = wf[CO + j];
#pragma unroll
    for (int p = 0; p < PPB; ++p) fa[p] = fmaf(wvv, comb[j][p], fa[p]);
  }

  float rv[PPB];
#pragma unroll
  for (int p = 0; p < PPB; ++p) rv[p] = 0.f;
  const float* wr = Wr + (size_t)c * CI;
  for (int i = 0; i < CI; ++i) {
    const float wvv = wr[i];
#pragma unroll
    for (int p = 0; p < PPB; ++p) rv[p] = fmaf(wvv, xv[i][p], rv[p]);
  }

  const float gg = g4[c], bb = b4[c], mm = m4[c], vv = v4[c], rsv = rs[0];
#pragma unroll
  for (int p = 0; p < PPB; ++p) {
    const float y = fmaxf(bn_f(fa[p], gg, bb, mm, vv), 0.f);
    out[base + p] = y + rsv * rv[p];
  }
}

// ---------------------------------------------------------------------------
extern "C" void kernel_launch(void* const* d_in, const int* in_sizes, int n_in,
                              void* d_out, int out_size, void* d_ws, size_t ws_size,
                              hipStream_t stream) {
  const float* x   = (const float*)d_in[0];
  const float* W3  = (const float*)d_in[1];
  const float* b3  = (const float*)d_in[2];
  const float* g1  = (const float*)d_in[3];
  const float* b1  = (const float*)d_in[4];
  const float* m1  = (const float*)d_in[5];
  const float* v1  = (const float*)d_in[6];
  const float* Wa  = (const float*)d_in[7];
  const float* ba  = (const float*)d_in[8];
  const float* g2  = (const float*)d_in[9];
  const float* b2  = (const float*)d_in[10];
  const float* m2  = (const float*)d_in[11];
  const float* v2  = (const float*)d_in[12];
  const float* Wq  = (const float*)d_in[13];
  const float* bq  = (const float*)d_in[14];
  const float* Wk  = (const float*)d_in[15];
  const float* bk  = (const float*)d_in[16];
  const float* Wv  = (const float*)d_in[17];
  const float* bv  = (const float*)d_in[18];
  const float* gam = (const float*)d_in[19];
  const float* Wg  = (const float*)d_in[20];
  const float* bg  = (const float*)d_in[21];
  const float* g3  = (const float*)d_in[22];
  const float* b3n = (const float*)d_in[23];
  const float* m3  = (const float*)d_in[24];
  const float* v3  = (const float*)d_in[25];
  const float* Wf  = (const float*)d_in[26];
  const float* bf  = (const float*)d_in[27];
  const float* g4  = (const float*)d_in[28];
  const float* b4  = (const float*)d_in[29];
  const float* m4  = (const float*)d_in[30];
  const float* v4  = (const float*)d_in[31];
  const float* Wr  = (const float*)d_in[32];
  const float* rs  = (const float*)d_in[33];
  float* out = (float*)d_out;

  // workspace layout (~29 MB)
  float* ws_local = (float*)d_ws;
  float* ws_a     = ws_local + (size_t)B_ * CO * NP;
  float* ws_o     = ws_a     + (size_t)B_ * CO * NP;
  _Float16* ws_q  = (_Float16*)(ws_o + (size_t)B_ * CO * NP);
  _Float16* ws_k  = ws_q + (size_t)B_ * CQ * NP;
  _Float16* ws_v  = ws_k + (size_t)B_ * CQ * NP;

  conv3_bn_relu<<<dim3(B_ * (CO / 32) * H_), dim3(W_), 0, stream>>>(
      x, W3, b3, g1, b1, m1, v1, ws_local);
  a_proj<<<dim3(B_ * (NP / PPB)), dim3(CO), 0, stream>>>(
      x, Wa, ba, g2, b2, m2, v2, ws_a);
  qkv_proj<<<dim3(B_ * (NP / PPB)), dim3(160), 0, stream>>>(
      ws_a, Wq, bq, Wk, bk, Wv, bv, ws_q, ws_k, ws_v);
  attn_flash<<<dim3(B_ * (NP / 128)), dim3(256), 0, stream>>>(
      ws_q, ws_k, ws_v, ws_o);
  fuse_out<<<dim3(B_ * (NP / PPB)), dim3(CO), 0, stream>>>(
      x, ws_local, ws_a, ws_o, gam, Wg, bg, g3, b3n, m3, v3,
      Wf, bf, g4, b4, m4, v4, Wr, rs, out);
}
